// LocalAutoCorr2DCUDA_51007031607352
// MI455X (gfx1250) — matrix operand order repaired and checked by reading, not compiled, not run on hardware
//
#include <hip/hip_runtime.h>
#include <hip/hip_bf16.h>

// ---------------------------------------------------------------------------
// Local windowed 2D autocorrelation for MI455X (gfx1250, wave32, WMMA).
//
// out[b,c,wy,wx,oy,ox] = sum_{i,j} W[i,j] * W[i+oy-4, j+ox-4]   (zero-filled)
//
// A = 8 vertical shifts of W (K=64 flat), B = 8 horizontal shifts of W
// (K=64 flat)  =>  (A x B^T)[a,b] = out[4 - a_shift, 4 + b_shift].
// One 16x16x64 f16 WMMA (two chained 16x16x32) computes the full 64-offset
// autocorrelation of TWO windows (diagonal blocks).
//
// Roofline: ~158 MB traffic -> ~6.8 us HBM floor; scalar fp32 VALU would be
// ~20 us compute-bound, so the f16 WMMA path (fp32 accumulate) rides the
// memory roofline. Input staged global->LDS via the Tensor Data Mover with
// LDS row padding (stride 132 floats); 5 zeroed guard rows make the A-side
// vertical-shift loads branch-free.
// ---------------------------------------------------------------------------

typedef __attribute__((ext_vector_type(16))) _Float16     v16h;
typedef __attribute__((ext_vector_type(8)))  float        v8f;
typedef __attribute__((ext_vector_type(4)))  unsigned int u32x4;
typedef __attribute__((ext_vector_type(8)))  int          i32x8;
typedef __attribute__((ext_vector_type(4)))  int          i32x4;

#define NH 31
#define NW 31
#define IMG_W 128
#define STRIP_STRIDE 132            // 128 data + 4 pad floats (bank spread)
#define GUARD_TOP 3                 // rows -3..-1 zeroed (A shift range [-3,4])
#define GUARD_BOT 2                 // rows 8..9 zeroed
#define STRIP_ROWS (8 + GUARD_TOP + GUARD_BOT)   // 13 rows

#if defined(__HIP_DEVICE_COMPILE__) && __has_builtin(__builtin_amdgcn_tensor_load_to_lds)
#define USE_TDM 1
#else
#define USE_TDM 0
#endif

__global__ __launch_bounds__(256) void
LocalAutoCorr2D_wmma_kernel(const float* __restrict__ x, float* __restrict__ out)
{
    // Row r of the image strip lives at (r + GUARD_TOP) * STRIP_STRIDE.
    __shared__ __align__(16) float strip[STRIP_ROWS * STRIP_STRIDE];

    const int       wy  = (int)(blockIdx.x % NH);
    const long long bc  = (long long)(blockIdx.x / NH);
    const float*    src = x + (bc * 128 + (long long)wy * 4) * IMG_W;   // 8 rows x 128

    // Zero the 5 guard rows (disjoint from the TDM-written data rows, so this
    // can proceed concurrently with the DMA; the barrier below orders both).
    for (int i = (int)threadIdx.x; i < (GUARD_TOP + GUARD_BOT) * STRIP_STRIDE; i += 256) {
        const int idx = (i < GUARD_TOP * STRIP_STRIDE) ? i : (8 * STRIP_STRIDE + i);
        strip[idx] = 0.0f;
    }

#if USE_TDM
    // --- Tensor Data Mover: DMA the 8x128 fp32 strip global->LDS rows 0..7,
    // padded to stride 132 floats. Issued once per workgroup by wave 0.
    if (threadIdx.x < 32) {
        const unsigned long long ga     = (unsigned long long)(const void*)src;
        const unsigned           ldsOff =
            (unsigned)(unsigned long long)(void*)&strip[GUARD_TOP * STRIP_STRIDE];

        u32x4 g0;
        g0[0] = 1u;                                         // count=1, user mode
        g0[1] = ldsOff;                                     // lds_addr (bytes)
        g0[2] = (unsigned)ga;                               // global_addr[31:0]
        g0[3] = ((unsigned)(ga >> 32) & 0x1FFFFFFu)         // global_addr[56:32]
                | (2u << 30);                               // type = 2 ("image")

        i32x8 g1;
        g1[0] = (2 << 16)      // data_size = 4 bytes
              | (1 << 20)      // pad_enable
              | (6 << 22)      // pad_interval: 128 dwords
              | (3 << 25);     // pad_amount:   4 dwords
        g1[1] = IMG_W << 16;   // tensor_dim0[15:0] = 128  (bits 63:48)
        g1[2] = 8 << 16;       // tensor_dim1[15:0] = 8    (bits 95:80)
        g1[3] = IMG_W << 16;   // tile_dim0 = 128          (bits 127:112)
        g1[4] = 8;             // tile_dim1 = 8            (bits 143:128)
        g1[5] = IMG_W;         // tensor_dim0_stride = 128 (bits 191:160)
        g1[6] = 0;
        g1[7] = 0;

        i32x4 z4 = {0, 0, 0, 0};
        i32x8 z8 = {0, 0, 0, 0, 0, 0, 0, 0};
        __builtin_amdgcn_tensor_load_to_lds(g0, g1, z4, z4, z8, 0);
#if __has_builtin(__builtin_amdgcn_s_wait_tensorcnt)
        __builtin_amdgcn_s_wait_tensorcnt(0);
#else
        asm volatile("s_wait_tensorcnt 0x0" ::: "memory");
#endif
    }
#else
    // Fallback: cooperative float4 staging (256 threads x 1 float4 = 8x128).
    {
        const int t = (int)threadIdx.x;
        const int r  = t >> 5;
        const int c4 = (t & 31) * 4;
        const float4 v = *reinterpret_cast<const float4*>(src + r * IMG_W + c4);
        *reinterpret_cast<float4*>(&strip[(r + GUARD_TOP) * STRIP_STRIDE + c4]) = v;
    }
#endif
    __syncthreads();

    const int lane = (int)(threadIdx.x & 31);
    const int wave = (int)(threadIdx.x >> 5);
    const long long outRow = (bc * NH + wy) * (long long)NW * 64;

    // 8 waves x 2 pairs = 16 pairs = windows 0..31 (31 is a masked phantom).
    #pragma unroll
    for (int qq = 0; qq < 2; ++qq) {
        const int q   = wave * 2 + qq;
        const int wx0 = 2 * q;
        const int wx1 = 2 * q + 1;

        // ---------------- A fragment: vertical shifts of W, K=64 -----------
        // wave32 16-bit A layout: lane m%16 = row; lanes<16 hold K{0..7,16..23},
        // lanes>=16 hold K{8..15,24..31} (per 32-chunk). Guard rows make all
        // loads unconditional (ds_load_b128 + cvt, no masks).
        const int mA   = lane & 15;
        const int hiA  = lane >> 4;
        const int aSh  = (mA & 7) - 3;                // vertical shift in [-3,4]
        const int wxm  = (mA >> 3) ? wx1 : wx0;
        const int x0m  = 4 * (wxm > 30 ? 30 : wxm);   // clamp phantom window

        v16h aF[2];
        #pragma unroll
        for (int c = 0; c < 2; ++c) {
            #pragma unroll
            for (int half = 0; half < 2; ++half) {
                const int rr = 4 * c + hiA + 2 * half + aSh;        // in [-3, 11]
                const float* rp = &strip[(rr + GUARD_TOP) * STRIP_STRIDE + x0m];
                const float4 f0 = *reinterpret_cast<const float4*>(rp);
                const float4 f1 = *reinterpret_cast<const float4*>(rp + 4);
                aF[c][half * 8 + 0] = (_Float16)f0.x;
                aF[c][half * 8 + 1] = (_Float16)f0.y;
                aF[c][half * 8 + 2] = (_Float16)f0.z;
                aF[c][half * 8 + 3] = (_Float16)f0.w;
                aF[c][half * 8 + 4] = (_Float16)f1.x;
                aF[c][half * 8 + 5] = (_Float16)f1.y;
                aF[c][half * 8 + 6] = (_Float16)f1.z;
                aF[c][half * 8 + 7] = (_Float16)f1.w;
            }
        }

        // ---------------- B fragment: horizontal shifts of W, K=64 ---------
        // B layout: lane n%16 = column; lanes<16 hold K{0..7,16..23}, lanes>=16 hold
        // K{8..15,24..31} (per 32-chunk). Loads are unconditional (clamped index);
        // window-boundary zero-fill is a value select (v_cndmask), no branch.
        const int nB  = lane & 15;
        const int hiB = lane >> 4;
        const int bSh = (nB & 7) - 4;                 // horizontal shift in [-4,3]
        const int wxn = (nB >> 3) ? wx1 : wx0;
        const int x0n = 4 * (wxn > 30 ? 30 : wxn);

        v16h bF[2];
        #pragma unroll
        for (int c = 0; c < 2; ++c) {
            #pragma unroll
            for (int half = 0; half < 2; ++half) {
                const int ii = 4 * c + hiB + 2 * half;              // in [0, 7]
                const float* rp = &strip[(ii + GUARD_TOP) * STRIP_STRIDE + x0n];
                #pragma unroll
                for (int j = 0; j < 8; ++j) {
                    const int  jb = j + bSh;                        // window-local col
                    const bool ok = (unsigned)jb < 8u;              // zero-fill outside
                    const int  jc = ok ? jb : 0;                    // clamped (in-range)
                    const float v = rp[jc];                         // unconditional load
                    bF[c][half * 8 + j] = (_Float16)(ok ? v : 0.0f);
                }
            }
        }

        // ---------------- 16x16x64 matmul: two chained WMMAs ---------------
        v8f acc = {};
        acc = __builtin_amdgcn_wmma_f32_16x16x32_f16(false, aF[0], false, bF[0],
                                                     (short)0, acc, false, false);
        acc = __builtin_amdgcn_wmma_f32_16x16x32_f16(false, aF[1], false, bF[1],
                                                     (short)0, acc, false, false);

        // ---------------- store diagonal blocks ----------------------------
        // D layout: lane L holds column n=L%16, vgpr r holds row m=r+8*(L/16).
        // Useful lanes: 0..7 (window wx0), 24..31 (window wx1).
        // D[m,n] = out[oy = 7 - (m%8), ox = n%8].
        if (lane < 8) {
            if (wx0 < NW) {
                float* o = out + outRow + (long long)wx0 * 64;
                #pragma unroll
                for (int r = 0; r < 8; ++r)
                    o[(7 - r) * 8 + lane] = acc[r];
            }
        } else if (lane >= 24) {
            if (wx1 < NW) {
                float* o = out + outRow + (long long)wx1 * 64;
                #pragma unroll
                for (int r = 0; r < 8; ++r)
                    o[(7 - r) * 8 + (lane - 24)] = acc[r];
            }
        }
    }
}

extern "C" void kernel_launch(void* const* d_in, const int* in_sizes, int n_in,
                              void* d_out, int out_size, void* d_ws, size_t ws_size,
                              hipStream_t stream)
{
    (void)in_sizes; (void)n_in; (void)d_ws; (void)ws_size; (void)out_size;
    const float* x   = (const float*)d_in[0];
    float*       out = (float*)d_out;

    // One block per (b, c, wy) row-strip: 8*64*31 = 15872 blocks, 256 threads
    // (8 wave32 waves), each wave computes 2 window-pairs via WMMA.
    const int blocks = 8 * 64 * NH;
    LocalAutoCorr2D_wmma_kernel<<<blocks, 256, 0, stream>>>(x, out);
}
